// ImprovedEEGModel_6519760355729
// MI455X (gfx1250) — matrix operand order repaired and checked by reading, not compiled, not run on hardware
//
#include <hip/hip_runtime.h>
#include <hip/hip_bf16.h>
#include <math.h>

// ---------------------------------------------------------------------------
// MI455X fused EEG model.
//  B=256, C=129, T=200.  conv2 (108 GFLOP) -> WMMA f16 GEMM, fused with
//  conv1 (VALU), BN folds, conv3 grouped reduction (register accumulators),
//  conv4/conv5/fcb in-LDS.  One workgroup (8 wave32) per batch sample so the
//  845MB conv2 activation never touches HBM.  x rows staged into LDS with
//  CDNA5 async global->LDS loads (ASYNCcnt).
// ---------------------------------------------------------------------------

typedef __attribute__((ext_vector_type(16))) _Float16 v16h;
typedef __attribute__((ext_vector_type(8)))  float    v8f;

#define Bx 256
#define Cc 129
#define Tt 200
#define INV_BN 0.99999500003749975f   // 1/sqrt(1+1e-5)

// ---- workspace layout (float offsets) -------------------------------------
#define WS_ATT_MEAN 0                       // B*C
#define WS_ATT_MAX  (WS_ATT_MEAN + Bx*Cc)   // B*C
#define WS_ATT      (WS_ATT_MAX  + Bx*Cc)   // B*C
#define WS_BACK     (WS_ATT      + Bx*Cc)   // B*64
#define WS_F64      (WS_BACK     + Bx*64)   // B*64
#define WS_FEATS    (WS_F64      + Bx*64)   // B*516
#define WS_FS2      (WS_FEATS    + Bx*516)  // 32
#define WS_FB2      (WS_FS2      + 32)      // 32
#define WS_AF16     (WS_FB2      + 32)      // 8192 halves (4096 floats), 32B aligned

// ---------------------------------------------------------------------------
// Prep: fold BN1 into conv2 weights, build fused post-GEMM scale/bias
// (BN2 + propagated BN1 bias), and swizzle W2' (32x256 f16) into a WMMA
// 16-bit A layout, contiguous k = 16*half + slot: tile(mt,kt) -> 32 lanes x 16 halves contiguous.
// ---------------------------------------------------------------------------
__global__ __launch_bounds__(256) void prep_k(const float* __restrict__ w2,
                                              const float* __restrict__ g1,
                                              const float* __restrict__ b1,
                                              const float* __restrict__ g2,
                                              const float* __restrict__ b2,
                                              float* fs2, float* fb2,
                                              _Float16* Ah) {
  const int tid = threadIdx.x;
  if (tid < 32) {
    const int o = tid;
    float bias2 = 0.f;
    for (int m = 0; m < 8; ++m) {
      const float bb = b1[m];
      for (int k = 0; k < 32; ++k) bias2 += w2[(o * 8 + m) * 32 + k] * bb;
    }
    const float s2 = g2[o] * INV_BN;
    fs2[o] = s2;
    fb2[o] = bias2 * s2 + b2[o];
  }
  // A layout: lanes 0-15 hold M=lane, K = slot (0..15);
  // lanes 16-31 same M with K = 16 + slot.
  for (int e = tid; e < 8192; e += 256) {
    const int slot = e & 15;
    const int lane = (e >> 4) & 31;
    const int tile = e >> 9;            // 0..15
    const int mt = tile >> 3, kt = tile & 7;
    const int hi = lane >> 4, M = lane & 15;
    const int v = slot >> 1, j = slot & 1;
    const int K = 16 * hi + slot;
    const int o = mt * 16 + M;          // output channel (0..31)
    const float s1 = g1[kt] * INV_BN;   // kt == mid-channel m
    Ah[e] = (_Float16)(w2[(o * 8 + kt) * 32 + K] * s1);
  }
}

// ---------------------------------------------------------------------------
// Channel attention: per-(b,c) mean/max over T (one wave32 per row).
// ---------------------------------------------------------------------------
__global__ __launch_bounds__(256) void att_reduce_k(const float* __restrict__ x,
                                                    float* mean, float* mx) {
  const int row = blockIdx.x * 8 + (threadIdx.x >> 5);
  const int lane = threadIdx.x & 31;
  if (row >= Bx * Cc) return;
  const float* p = x + (size_t)row * Tt;
  float s = 0.f, m = -1e30f;
  for (int i = lane; i < Tt; i += 32) { float v = p[i]; s += v; m = fmaxf(m, v); }
  for (int off = 16; off; off >>= 1) {
    s += __shfl_xor(s, off);
    m = fmaxf(m, __shfl_xor(m, off));
  }
  if (lane == 0) { mean[row] = s * (1.f / Tt); mx[row] = m; }
}

__global__ __launch_bounds__(192) void att_fc_k(const float* __restrict__ mean,
                                                const float* __restrict__ mx,
                                                const float* __restrict__ w1,
                                                const float* __restrict__ w2,
                                                float* att) {
  const int b = blockIdx.x, tid = threadIdx.x;
  __shared__ float hs[16];
  const float* mr = mean + (size_t)b * Cc;
  const float* xr = mx + (size_t)b * Cc;
  if (tid < 16) {
    float a = 0.f, bs = 0.f;
    for (int c = 0; c < Cc; ++c) {
      const float w = w1[tid * Cc + c];
      a += w * mr[c]; bs += w * xr[c];
    }
    hs[tid] = fmaxf(a, 0.f) + fmaxf(bs, 0.f);   // relu(mean path)+relu(max path)
  }
  __syncthreads();
  if (tid < Cc) {
    float t = 0.f;
    for (int r = 0; r < 16; ++r) t += w2[tid * 16 + r] * hs[r];
    att[(size_t)b * Cc + tid] = 1.f / (1.f + __expf(-t));
  }
}

// ---------------------------------------------------------------------------
// Fused conv1 -> (BN1) -> conv2(WMMA) -> (BN2+att) -> conv3 -> BN3/ELU/pool ->
// conv4/BN4 -> conv5/BN5/ELU/pool -> fcb.  One block per batch sample b.
// ---------------------------------------------------------------------------
__global__ __launch_bounds__(256) void convnet_k(const float* __restrict__ x,
                                                 const float* __restrict__ att,
                                                 const float* __restrict__ w1g,
                                                 const float* __restrict__ w3,
                                                 const float* __restrict__ w4,
                                                 const float* __restrict__ w5,
                                                 const float* __restrict__ g3,
                                                 const float* __restrict__ b3,
                                                 const float* __restrict__ g4,
                                                 const float* __restrict__ b4,
                                                 const float* __restrict__ g5,
                                                 const float* __restrict__ b5,
                                                 const float* __restrict__ fcbw,
                                                 const float* __restrict__ fcbb,
                                                 const float* __restrict__ fs2g,
                                                 const float* __restrict__ fb2g,
                                                 const _Float16* __restrict__ Ag,
                                                 float* __restrict__ back) {
  const int b = blockIdx.x;
  const int tid = threadIdx.x;
  const int lane = tid & 31;
  const int wave = tid >> 5;

  __shared__ __align__(32) _Float16 Alds[8192];   // W2' fp16, A-matrix layout
  __shared__ float w1s[256];                      // conv1 weights (8x32)
  __shared__ float fs2s[32], fb2s[32];
  __shared__ float xr[Tt];                        // raw x row (att folded later)
  __shared__ _Float16 midh[8][240];               // conv1 output, conv2-padded
  __shared__ float w3c[64];                       // conv3 column c
  __shared__ float h3[64][Tt];                    // conv3 out (post BN3+ELU)
  __shared__ float p3[64][52];                    // pooled
  __shared__ float h4[32][52];
  __shared__ float h5s[8][52];
  __shared__ float feat96[96];

  { // stage constants
    const uint* Agu = (const uint*)Ag;
    uint* Au = (uint*)Alds;
    for (int i = tid; i < 4096; i += 256) Au[i] = Agu[i];
    for (int i = tid; i < 256; i += 256) w1s[i] = w1g[i];
    if (tid < 32) { fs2s[tid] = fs2g[tid]; fb2s[tid] = fb2g[tid]; }
  }

  // conv3 accumulators: fully-unrolled tile slots so clang promotes them to
  // VGPRs (no scratch spill).  Each lane's GEMM D element (o,t) feeds conv3
  // channels 2o / 2o+1 exclusively, so no cross-lane reduction is needed.
  float acc3r[4][16];
  #pragma unroll
  for (int s = 0; s < 4; ++s)
    #pragma unroll
    for (int j = 0; j < 16; ++j) acc3r[s][j] = 0.f;

  for (int c = 0; c < Cc; ++c) {
    const float a = att[(size_t)b * Cc + c];     // folded into BN2 below
    const float* xrow = x + ((size_t)b * Cc + c) * Tt;
    // Async global->LDS copy of the raw row (pure copy now that the attention
    // scale is folded downstream).  Tracked with ASYNCcnt.
    if (tid < Tt) {
      const unsigned ldsa = (unsigned)(size_t)(&xr[tid]);
      asm volatile("global_load_async_to_lds_b32 %0, %1, %2"
                   :: "v"(ldsa), "v"(tid * 4), "s"(xrow)
                   : "memory");
    }
    if (tid < 64) w3c[tid] = w3[tid * Cc + c];
    if (c + 1 < Cc && lane == 0)
      __builtin_prefetch(x + ((size_t)b * Cc + c + 1) * Tt, 0, 1);
    asm volatile("s_wait_asynccnt 0x0" ::: "memory");
    __syncthreads();

    // conv1 (raw; BN1 folded into W2'): mid positions -15..224 (stored +15),
    // zero outside conv1's valid range 0..199 (== conv2's zero padding).
    for (int e = tid; e < 8 * 240; e += 256) {
      const int m = e / 240, pos = e - m * 240, pp = pos - 15;
      float v = 0.f;
      if (pp >= 0 && pp < Tt) {
        #pragma unroll
        for (int k = 0; k < 32; ++k) {
          const int xi = pp + k - 15;
          if (xi >= 0 && xi < Tt) v += w1s[m * 32 + k] * xr[xi];
        }
      }
      midh[m][pos] = (_Float16)v;
    }
    __syncthreads();

    // conv2 GEMM: D[32,208] = W2'[32,256] x col[256,208]; K-tile kt == mid
    // channel m, so B operand is a contiguous shifted window of midh[kt].
    #pragma unroll
    for (int i = 0; i < 4; ++i) {
      const int tt = wave + 8 * i;                // wave-uniform guard
      if (tt < 26) {
        const int mt = tt / 13, nt = tt % 13;
        v8f d;
        #pragma unroll
        for (int r = 0; r < 8; ++r) d[r] = 0.f;
        #pragma unroll
        for (int kt = 0; kt < 8; ++kt) {
          const v16h av = *(const v16h*)&Alds[(((mt << 3) | kt) * 32 + lane) << 4];
          v16h bv;
          const int n0 = lane & 15;
          const int basei = nt * 16 + n0 + ((lane >> 4) << 4);
          #pragma unroll
          for (int v2 = 0; v2 < 16; ++v2) bv[v2] = midh[kt][basei + v2];
          d = __builtin_amdgcn_wmma_f32_16x16x32_f16(false, av, false, bv,
                                                     (short)0, d, false, false);
        }
        // BN2 (+attention scale) + conv3 fold into register accumulators
        const int hi = lane >> 4, n = lane & 15, t = nt * 16 + n;
        if (t < Tt) {
          #pragma unroll
          for (int r = 0; r < 8; ++r) {
            const int o = mt * 16 + r + 8 * hi;
            const float val = d[r] * (fs2s[o] * a) + fb2s[o];
            acc3r[i][2 * r]     += w3c[2 * o] * val;
            acc3r[i][2 * r + 1] += w3c[2 * o + 1] * val;
          }
        }
      }
    }
    __syncthreads();
  }

  // BN3 + ELU -> h3
  #pragma unroll
  for (int i = 0; i < 4; ++i) {
    const int tt = wave + 8 * i;
    if (tt < 26) {
      const int mt = tt / 13, nt = tt % 13;
      const int hi = lane >> 4, n = lane & 15, t = nt * 16 + n;
      if (t < Tt) {
        #pragma unroll
        for (int r = 0; r < 8; ++r) {
          const int o = mt * 16 + r + 8 * hi;
          #pragma unroll
          for (int j = 0; j < 2; ++j) {
            const int oc = 2 * o + j;
            float v = acc3r[i][2 * r + j] * (g3[oc] * INV_BN) + b3[oc];
            h3[oc][t] = v > 0.f ? v : (__expf(v) - 1.f);
          }
        }
      }
    }
  }
  __syncthreads();

  for (int e = tid; e < 64 * 50; e += 256) {   // avgpool4
    const int oc = e / 50, q = e - oc * 50;
    p3[oc][q] = 0.25f * (h3[oc][4 * q] + h3[oc][4 * q + 1] +
                         h3[oc][4 * q + 2] + h3[oc][4 * q + 3]);
  }
  __syncthreads();

  for (int e = tid; e < 32 * 50; e += 256) {   // conv4 (dil=2, pad 15) + BN4
    const int oc = e / 50, t = e - oc * 50;
    float s = 0.f;
    for (int ic = 0; ic < 64; ++ic) {
      const float* wr = w4 + (oc * 64 + ic) * 16;
      #pragma unroll
      for (int k = 0; k < 16; ++k) {
        const int ti = t + 2 * k - 15;
        if (ti >= 0 && ti < 50) s += wr[k] * p3[ic][ti];
      }
    }
    h4[oc][t] = s * (g4[oc] * INV_BN) + b4[oc];
  }
  __syncthreads();

  for (int e = tid; e < 8 * 50; e += 256) {    // conv5 (dil=4, pad 30) + BN5 + ELU
    const int oc = e / 50, t = e - oc * 50;
    float s = 0.f;
    for (int ic = 0; ic < 32; ++ic) {
      const float* wr = w5 + (oc * 32 + ic) * 16;
      #pragma unroll
      for (int k = 0; k < 16; ++k) {
        const int ti = t + 4 * k - 30;
        if (ti >= 0 && ti < 50) s += wr[k] * h4[ic][ti];
      }
    }
    const float v = s * (g5[oc] * INV_BN) + b5[oc];
    h5s[oc][t] = v > 0.f ? v : (__expf(v) - 1.f);
  }
  __syncthreads();

  if (tid < 96) {                              // avgpool4 -> flatten (8x12)
    const int oc = tid / 12, q = tid % 12;
    feat96[tid] = 0.25f * (h5s[oc][4 * q] + h5s[oc][4 * q + 1] +
                           h5s[oc][4 * q + 2] + h5s[oc][4 * q + 3]);
  }
  __syncthreads();

  if (tid < 64) {                              // fcb
    float s = fcbb[tid];
    for (int i = 0; i < 96; ++i) s += fcbw[tid * 96 + i] * feat96[i];
    back[(size_t)b * 64 + tid] = s;
  }
}

// ---------------------------------------------------------------------------
// Welch band power (on raw x): 5 Hann segments, rFFT bins 3..31 only (the
// bins the 4 bands touch), sum(win^2)=24 -> scale = 1/2400, interior x2.
// ---------------------------------------------------------------------------
__global__ __launch_bounds__(64) void welch_k(const float* __restrict__ x,
                                              float* __restrict__ feats) {
  const int row = blockIdx.x;
  const int b = row / Cc, c = row % Cc;
  const int tid = threadIdx.x;
  __shared__ float xw[Tt];
  __shared__ float ms[5];
  __shared__ float bs[4];
  for (int i = tid; i < Tt; i += 64) xw[i] = x[((size_t)b * Cc + c) * Tt + i];
  if (tid < 4) bs[tid] = 0.f;
  __syncthreads();
  if (tid < 5) {
    float s = 0.f;
    for (int n = 0; n < 64; ++n) s += xw[tid * 32 + n];
    ms[tid] = s * (1.f / 64.f);
  }
  __syncthreads();
  for (int task = tid; task < 145; task += 64) {     // 5 segs x 29 bins
    const int s = task / 29, bi = task - 29 * s, k = bi + 3;
    float re = 0.f, im = 0.f;
    const float base = 6.2831853071795864f / 64.f;
    for (int n = 0; n < 64; ++n) {
      const float w = 0.5f - 0.5f * __cosf(base * n);
      const float v = (xw[s * 32 + n] - ms[s]) * w;
      float sn, cs;
      __sincosf(base * (float)(k * n), &sn, &cs);
      re += v * cs;
      im -= v * sn;
    }
    const float psd = (re * re + im * im) * (2.f / 2400.f);
    const int band = (k <= 5) ? 0 : (k <= 8) ? 1 : (k <= 19) ? 2 : 3;
    atomicAdd(&bs[band], psd);
  }
  __syncthreads();
  if (tid < 4) {
    const float cnt[4] = {3.f, 3.f, 11.f, 12.f};
    feats[(size_t)b * (Cc * 4) + c * 4 + tid] = bs[tid] / (5.f * cnt[tid]);
  }
}

// ---------------------------------------------------------------------------
// Frequency-feature MLP and head.
// ---------------------------------------------------------------------------
__global__ __launch_bounds__(128) void fe_k(const float* __restrict__ feats,
                                            const float* __restrict__ w1,
                                            const float* __restrict__ b1,
                                            const float* __restrict__ w2,
                                            const float* __restrict__ b2,
                                            float* __restrict__ f64o) {
  const int b = blockIdx.x, tid = threadIdx.x;
  __shared__ float fl[516];
  __shared__ float h1[128];
  for (int i = tid; i < 516; i += 128) fl[i] = feats[(size_t)b * 516 + i];
  __syncthreads();
  float s = b1[tid];
  for (int i = 0; i < 516; ++i) s += w1[tid * 516 + i] * fl[i];
  h1[tid] = fmaxf(s, 0.f);
  __syncthreads();
  if (tid < 64) {
    float t = b2[tid];
    for (int i = 0; i < 128; ++i) t += w2[tid * 128 + i] * h1[i];
    f64o[(size_t)b * 64 + tid] = fmaxf(t, 0.f);
  }
}

__global__ __launch_bounds__(128) void head_k(const float* __restrict__ back,
                                              const float* __restrict__ f64i,
                                              const float* __restrict__ w1,
                                              const float* __restrict__ b1,
                                              const float* __restrict__ w2,
                                              const float* __restrict__ b2,
                                              const float* __restrict__ w3,
                                              const float* __restrict__ b3,
                                              float* __restrict__ out) {
  const int b = blockIdx.x, tid = threadIdx.x;
  __shared__ float z[128];
  __shared__ float z1[128];
  __shared__ float z2[64];
  if (tid < 64) {
    z[tid] = back[(size_t)b * 64 + tid];
    z[64 + tid] = f64i[(size_t)b * 64 + tid];
  }
  __syncthreads();
  float s = b1[tid];
  for (int i = 0; i < 128; ++i) s += w1[tid * 128 + i] * z[i];
  z1[tid] = fmaxf(s, 0.f);
  __syncthreads();
  if (tid < 64) {
    float t = b2[tid];
    for (int i = 0; i < 128; ++i) t += w2[tid * 128 + i] * z1[i];
    z2[tid] = fmaxf(t, 0.f);
  }
  __syncthreads();
  if (tid == 0) {
    float t = b3[0];
    for (int i = 0; i < 64; ++i) t += w3[i] * z2[i];
    out[b] = t;
  }
}

// ---------------------------------------------------------------------------
extern "C" void kernel_launch(void* const* d_in, const int* in_sizes, int n_in,
                              void* d_out, int out_size, void* d_ws, size_t ws_size,
                              hipStream_t stream) {
  // setup_inputs order: x, then params in dict-insertion order.
  const float* x      = (const float*)d_in[0];
  const float* ca_w1  = (const float*)d_in[1];
  const float* ca_w2  = (const float*)d_in[2];
  const float* conv1w = (const float*)d_in[3];
  const float* bn1g   = (const float*)d_in[4];
  const float* bn1b   = (const float*)d_in[5];
  const float* conv2w = (const float*)d_in[6];
  const float* bn2g   = (const float*)d_in[7];
  const float* bn2b   = (const float*)d_in[8];
  const float* conv3w = (const float*)d_in[9];
  const float* bn3g   = (const float*)d_in[10];
  const float* bn3b   = (const float*)d_in[11];
  const float* conv4w = (const float*)d_in[12];
  const float* bn4g   = (const float*)d_in[13];
  const float* bn4b   = (const float*)d_in[14];
  const float* conv5w = (const float*)d_in[15];
  const float* bn5g   = (const float*)d_in[16];
  const float* bn5b   = (const float*)d_in[17];
  const float* fcbw   = (const float*)d_in[18];
  const float* fcbb   = (const float*)d_in[19];
  const float* fe_w1  = (const float*)d_in[20];
  const float* fe_b1  = (const float*)d_in[21];
  const float* fe_w2  = (const float*)d_in[22];
  const float* fe_b2  = (const float*)d_in[23];
  const float* h_w1   = (const float*)d_in[24];
  const float* h_b1   = (const float*)d_in[25];
  const float* h_w2   = (const float*)d_in[26];
  const float* h_b2   = (const float*)d_in[27];
  const float* h_w3   = (const float*)d_in[28];
  const float* h_b3   = (const float*)d_in[29];

  float* ws = (float*)d_ws;
  float* att_mean = ws + WS_ATT_MEAN;
  float* att_max  = ws + WS_ATT_MAX;
  float* att      = ws + WS_ATT;
  float* back     = ws + WS_BACK;
  float* f64      = ws + WS_F64;
  float* feats    = ws + WS_FEATS;
  float* fs2      = ws + WS_FS2;
  float* fb2      = ws + WS_FB2;
  _Float16* Ah    = (_Float16*)(ws + WS_AF16);

  prep_k<<<1, 256, 0, stream>>>(conv2w, bn1g, bn1b, bn2g, bn2b, fs2, fb2, Ah);
  att_reduce_k<<<(Bx * Cc + 7) / 8, 256, 0, stream>>>(x, att_mean, att_max);
  att_fc_k<<<Bx, 192, 0, stream>>>(att_mean, att_max, ca_w1, ca_w2, att);
  convnet_k<<<Bx, 256, 0, stream>>>(x, att, conv1w, conv3w, conv4w, conv5w,
                                    bn3g, bn3b, bn4g, bn4b, bn5g, bn5b,
                                    fcbw, fcbb, fs2, fb2, Ah, back);
  welch_k<<<Bx * Cc, 64, 0, stream>>>(x, feats);
  fe_k<<<Bx, 128, 0, stream>>>(feats, fe_w1, fe_b1, fe_w2, fe_b2, f64);
  head_k<<<Bx, 128, 0, stream>>>(back, f64, h_w1, h_b1, h_w2, h_b2, h_w3, h_b3,
                                 (float*)d_out);
}
